// Qwen3NextAttentionForEngine_26156350832953
// MI455X (gfx1250) — compile-verified, matrix operand order checked by reading, not run on hardware
//
#include <hip/hip_runtime.h>
#include <hip/hip_bf16.h>
#include <math.h>

// Problem constants (Qwen3-Next attention, from reference)
#define BB   2
#define SS   2048
#define HID  2048
#define NH   16
#define NKV  4
#define HD   128
#define MTOK (BB*SS)        // 4096 tokens
#define QGN  (2*NH*HD)      // 4096 (q + gate)
#define KVN  (NKV*HD)       // 512
#define KTILE 64

typedef unsigned short u16;
typedef unsigned int   u32;

typedef __attribute__((ext_vector_type(16))) __bf16 bf16x16;
typedef __attribute__((ext_vector_type(8)))  float  f32x8;
typedef __attribute__((ext_vector_type(4)))  int    i32x4;

union Frag {
  uint4   u[2];   // 32 bytes
  bf16x16 v;
};

__device__ __forceinline__ u16 f2bf(float f) {
  u32 u = __builtin_bit_cast(u32, f);
  u32 r = u + 0x7FFFu + ((u >> 16) & 1u);   // round-to-nearest-even
  return (u16)(r >> 16);
}

__device__ __forceinline__ f32x8 wmma_bf16(bf16x16 a, bf16x16 b, f32x8 c) {
  // (neg_a, A, neg_b, B, c_mod, C, reuse_a, reuse_b)
  return __builtin_amdgcn_wmma_f32_16x16x32_bf16(false, a, false, b, (short)0, c, false, false);
}

// ---------------------------------------------------------------------------
// Async global -> LDS staging (CDNA5 ASYNCcnt path), with safe fallback.
// Builtin signature (from clang diagnostic): (i32x4 AS1*, i32x4 AS3*, imm, imm)
// ---------------------------------------------------------------------------
#if __has_builtin(__builtin_amdgcn_global_load_async_to_lds_b128) && \
    __has_builtin(__builtin_amdgcn_s_wait_asynccnt)
#define HAVE_ASYNC_LDS 1
#else
#define HAVE_ASYNC_LDS 0
#endif

typedef __attribute__((address_space(1))) i32x4* gptr128_t;
typedef __attribute__((address_space(3))) i32x4* lptr128_t;

__device__ __forceinline__ void stage_b128(const u16* gp, u16* lp) {
#if HAVE_ASYNC_LDS
  i32x4* g = (i32x4*)const_cast<u16*>(gp);
  i32x4* l = (i32x4*)lp;
  __builtin_amdgcn_global_load_async_to_lds_b128((gptr128_t)g, (lptr128_t)l, 0, 0);
#else
  *(uint4*)lp = *(const uint4*)gp;
#endif
}

__device__ __forceinline__ void stage_wait() {
#if HAVE_ASYNC_LDS
  __builtin_amdgcn_s_wait_asynccnt(0);
#endif
}

// ---------------------------------------------------------------------------
// f32 -> bf16 cast (grid-stride)
// ---------------------------------------------------------------------------
__global__ void cast_kernel(const float* __restrict__ in, u16* __restrict__ out, size_t n) {
  size_t stride = (size_t)gridDim.x * blockDim.x;
  for (size_t i = (size_t)blockIdx.x * blockDim.x + threadIdx.x; i < n; i += stride)
    out[i] = f2bf(in[i]);
}

// ---------------------------------------------------------------------------
// C[M,N] (f32) = A[M,K] (bf16, row-major) * W[N,K]^T (bf16, row-major weights)
// Block tile 128x128, 8 waves (256 thr), wave = 32(M) x 64(N).
// K-tile 64, double-buffered LDS filled by async global->LDS DMA: iteration i
// issues DMA for tile i+1 while WMMAs consume tile i.
// ---------------------------------------------------------------------------
__global__ __launch_bounds__(256) void gemm_kernel(
    const u16* __restrict__ A, const u16* __restrict__ W, float* __restrict__ C,
    int M, int N, int K)
{
  __shared__ u16 lA[2][128 * KTILE];
  __shared__ u16 lW[2][128 * KTILE];

  int tid  = threadIdx.x;
  int lane = tid & 31;
  int l16  = lane & 15;
  int hi   = lane >> 4;
  int kb8  = hi * 8;
  int wid  = tid >> 5;
  int wm   = wid & 3;       // 4 waves along M
  int wn   = wid >> 2;      // 2 waves along N
  int mbase = blockIdx.y * 128;
  int nbase = blockIdx.x * 128;

  f32x8 acc[2][4] = {};

  // Stage one 128xKTILE tile of A and W into LDS buffer `buf` (16B per lane-op)
  auto stage = [&](int buf, int kt) {
    #pragma unroll
    for (int t = 0; t < 4; ++t) {
      int li  = tid + t * 256;          // 0..1023
      int row = li >> 3;                // 128 rows
      int c8  = (li & 7) * 8;           // 8 chunks of 8 bf16 per row
      stage_b128(&A[(size_t)(mbase + row) * K + kt + c8], &lA[buf][row * KTILE + c8]);
      stage_b128(&W[(size_t)(nbase + row) * K + kt + c8], &lW[buf][row * KTILE + c8]);
    }
  };

  stage(0, 0);
  stage_wait();
  __syncthreads();

  int nk = K / KTILE;
  for (int it = 0; it < nk; ++it) {
    int buf = it & 1;
    if (it + 1 < nk) stage(buf ^ 1, (it + 1) * KTILE);   // async prefetch next tile

    #pragma unroll
    for (int ks = 0; ks < KTILE; ks += 32) {
      Frag fa[2];
      #pragma unroll
      for (int mt = 0; mt < 2; ++mt) {
        const u16* p = &lA[buf][(wm * 32 + mt * 16 + l16) * KTILE + ks + kb8];
        fa[mt].u[0] = *(const uint4*)(p);
        fa[mt].u[1] = *(const uint4*)(p + 16);
      }
      #pragma unroll
      for (int nt = 0; nt < 4; ++nt) {
        Frag fb;
        const u16* p = &lW[buf][(wn * 64 + nt * 16 + l16) * KTILE + ks + kb8];
        fb.u[0] = *(const uint4*)(p);
        fb.u[1] = *(const uint4*)(p + 16);
        #pragma unroll
        for (int mt = 0; mt < 2; ++mt)
          acc[mt][nt] = wmma_bf16(fa[mt].v, fb.v, acc[mt][nt]);
      }
    }

    stage_wait();          // next tile fully landed in LDS
    __syncthreads();       // all waves done reading current tile
  }

  #pragma unroll
  for (int mt = 0; mt < 2; ++mt)
    #pragma unroll
    for (int nt = 0; nt < 4; ++nt)
      #pragma unroll
      for (int r = 0; r < 8; ++r) {
        size_t row = (size_t)(mbase + wm * 32 + mt * 16 + r + 8 * hi);
        size_t col = (size_t)(nbase + wn * 64 + nt * 16 + l16);
        C[row * (size_t)N + col] = acc[mt][nt][r];
      }
}

// ---------------------------------------------------------------------------
// Per-head RMSNorm + RoPE; repack Q->(b,h,s,d), K->(b,hk,s,d), V->(b,hk,d,s) bf16
// One wave per (token, head-task); 20 tasks (16 Q + 4 KV) per token, 4 waves/blk.
// ---------------------------------------------------------------------------
__global__ __launch_bounds__(128) void normrope_kernel(
    const float* __restrict__ qg, const float* __restrict__ kf, const float* __restrict__ vf,
    const float* __restrict__ cs, const float* __restrict__ sn,
    const float* __restrict__ qw, const float* __restrict__ kw,
    u16* __restrict__ qb, u16* __restrict__ kb, u16* __restrict__ vt)
{
  int t    = blockIdx.x;          // token
  int b    = t >> 11;
  int s    = t & (SS - 1);
  int w    = threadIdx.x >> 5;
  int lane = threadIdx.x & 31;

  for (int task = w; task < NH + NKV; task += 4) {
    bool isQ = task < NH;
    int h = isQ ? task : task - NH;
    const float* src = isQ ? &qg[(size_t)t * QGN + h * HD]
                           : &kf[(size_t)t * KVN + h * HD];
    const float* wgt = isQ ? qw : kw;

    float x[4];
    float ss = 0.f;
    #pragma unroll
    for (int j = 0; j < 4; ++j) { x[j] = src[j * 32 + lane]; ss += x[j] * x[j]; }
    #pragma unroll
    for (int off = 16; off >= 1; off >>= 1) ss += __shfl_xor(ss, off);
    float rs = rsqrtf(ss * (1.0f / HD) + 1e-6f);

    float xn[4];
    #pragma unroll
    for (int j = 0; j < 4; ++j) xn[j] = x[j] * rs * wgt[j * 32 + lane];

    // RoPE: element i pairs with i^64; i = j*32+lane -> partner is j^2, same lane
    u16 outv[4];
    #pragma unroll
    for (int j = 0; j < 4; ++j) {
      int i  = j * 32 + lane;
      float c  = cs[(size_t)t * HD + i];
      float si = sn[(size_t)t * HD + i];
      float rot = (j < 2) ? -xn[j ^ 2] : xn[j ^ 2];
      outv[j] = f2bf(xn[j] * c + rot * si);
    }

    if (isQ) {
      u16* dst = &qb[(((size_t)(b * NH + h)) * SS + s) * HD];
      #pragma unroll
      for (int j = 0; j < 4; ++j) dst[j * 32 + lane] = outv[j];
    } else {
      u16* dst = &kb[(((size_t)(b * NKV + h)) * SS + s) * HD];
      #pragma unroll
      for (int j = 0; j < 4; ++j) dst[j * 32 + lane] = outv[j];
      // V: plain cast + transpose to (b,hk,d,s)
      #pragma unroll
      for (int j = 0; j < 4; ++j) {
        int i = j * 32 + lane;
        float v = vf[(size_t)t * KVN + h * HD + i];
        vt[(((size_t)(b * NKV + h)) * HD + i) * SS + s] = f2bf(v);
      }
    }
  }
}

// ---------------------------------------------------------------------------
// Flash attention (causal, GQA G=4). Block = 4 waves; wave owns 16 query rows.
// Scores via WMMA (K-dim = head_dim 128), online softmax f32, P repacked via
// per-wave LDS into A-fragment layout, PV via WMMA against transposed V.
// K/V are L2-resident (8 MB total) -> direct global->VGPR fragment loads.
// ---------------------------------------------------------------------------
__global__ __launch_bounds__(128) void attn_kernel(
    const u16* __restrict__ Qb, const u16* __restrict__ Kb, const u16* __restrict__ Vt,
    float* __restrict__ Of)
{
  __shared__ u16 pbuf[4][16 * 32];
  const float SCALE = 0.08838834764831845f;   // 1/sqrt(128)
  const float L2E   = 1.4426950408889634f;

  int bh = blockIdx.x;
  int b = bh >> 4, h = bh & 15;
  int hk = h >> 2;
  int w    = threadIdx.x >> 5;
  int lane = threadIdx.x & 31;
  int l16  = lane & 15;
  int hi   = lane >> 4;
  int kb8  = hi * 8;
  int q0   = blockIdx.y * 64 + w * 16;

  const u16* Qh = Qb + ((size_t)(b * NH  + h )) * SS * HD;
  const u16* Kh = Kb + ((size_t)(b * NKV + hk)) * SS * HD;
  const u16* Vh = Vt + ((size_t)(b * NKV + hk)) * HD * SS;

  // Q fragments: 4 K-steps of 32 over head_dim
  Frag qf[4];
  #pragma unroll
  for (int s4 = 0; s4 < 4; ++s4) {
    const u16* p = &Qh[(size_t)(q0 + l16) * HD + s4 * 32 + kb8];
    qf[s4].u[0] = *(const uint4*)(p);
    qf[s4].u[1] = *(const uint4*)(p + 16);
  }

  f32x8 o[8] = {};
  float rm[8], ll[8];
  #pragma unroll
  for (int r = 0; r < 8; ++r) { rm[r] = -1e30f; ll[r] = 0.f; }

  int kbEnd = (q0 + 15) >> 5;
  for (int kblk = 0; kblk <= kbEnd; ++kblk) {
    int k0 = kblk * 32;

    // ---- scores: two 16x16 tiles (keys k0..+15, k0+16..+31) ----
    f32x8 s0 = {}, s1 = {};
    #pragma unroll
    for (int s4 = 0; s4 < 4; ++s4) {
      Frag f0, f1;
      const u16* p0 = &Kh[(size_t)(k0 + l16) * HD + s4 * 32 + kb8];
      f0.u[0] = *(const uint4*)(p0);
      f0.u[1] = *(const uint4*)(p0 + 16);
      const u16* p1 = p0 + 16 * HD;
      f1.u[0] = *(const uint4*)(p1);
      f1.u[1] = *(const uint4*)(p1 + 16);
      s0 = wmma_bf16(qf[s4].v, f0.v, s0);
      s1 = wmma_bf16(qf[s4].v, f1.v, s1);
    }

    // ---- online softmax (rows live across VGPR index r; cols across 16 lanes)
    float al[8];
    #pragma unroll
    for (int r = 0; r < 8; ++r) {
      int m = q0 + r + 8 * hi;
      float x0 = s0[r] * SCALE;
      float x1 = s1[r] * SCALE;
      if (k0 + l16 > m)      x0 = -1e30f;
      if (k0 + 16 + l16 > m) x1 = -1e30f;
      float mx = fmaxf(x0, x1);
      #pragma unroll
      for (int off = 1; off < 16; off <<= 1) mx = fmaxf(mx, __shfl_xor(mx, off));
      float nm = fmaxf(rm[r], mx);
      float alpha = exp2f((rm[r] - nm) * L2E);
      float p0 = exp2f((x0 - nm) * L2E);
      float p1 = exp2f((x1 - nm) * L2E);
      float sr = p0 + p1;
      #pragma unroll
      for (int off = 1; off < 16; off <<= 1) sr += __shfl_xor(sr, off);
      ll[r] = ll[r] * alpha + sr;
      rm[r] = nm;
      al[r] = alpha;
      s0[r] = p0;
      s1[r] = p1;
    }
    #pragma unroll
    for (int dt = 0; dt < 8; ++dt)
      #pragma unroll
      for (int r = 0; r < 8; ++r)
        o[dt][r] *= al[r];

    // ---- repack P (C-layout) -> A-fragment via per-wave LDS (same-wave DS in-order)
    u16* pb = pbuf[w];
    #pragma unroll
    for (int r = 0; r < 8; ++r) {
      int row = r + 8 * hi;
      pb[row * 32 + l16]      = f2bf(s0[r]);
      pb[row * 32 + 16 + l16] = f2bf(s1[r]);
    }
    Frag pf;
    pf.u[0] = *(const uint4*)&pb[l16 * 32 + kb8];
    pf.u[1] = *(const uint4*)&pb[l16 * 32 + 16 + kb8];

    // ---- PV: O[16 x 128] += P[16 x 32] @ V[32 x 128] (V pre-transposed (d,s))
    #pragma unroll
    for (int dt = 0; dt < 8; ++dt) {
      Frag vfr;
      const u16* vp = &Vh[(size_t)(dt * 16 + l16) * SS + k0 + kb8];
      vfr.u[0] = *(const uint4*)(vp);
      vfr.u[1] = *(const uint4*)(vp + 16);
      o[dt] = wmma_bf16(pf.v, vfr.v, o[dt]);
    }
  }

  #pragma unroll
  for (int dt = 0; dt < 8; ++dt)
    #pragma unroll
    for (int r = 0; r < 8; ++r) {
      int m = q0 + r + 8 * hi;
      Of[((size_t)b * SS + m) * HID + h * HD + dt * 16 + l16] = o[dt][r] / ll[r];
    }
}

// ---------------------------------------------------------------------------
// attn * sigmoid(gate) -> bf16
// ---------------------------------------------------------------------------
__global__ void gate_kernel(const float* __restrict__ attn, const float* __restrict__ qg,
                            u16* __restrict__ out)
{
  size_t total  = (size_t)MTOK * HID;
  size_t stride = (size_t)gridDim.x * blockDim.x;
  for (size_t i = (size_t)blockIdx.x * blockDim.x + threadIdx.x; i < total; i += stride) {
    size_t t = i >> 11;              // token
    int oo   = (int)(i & (HID - 1));
    float g  = qg[t * QGN + HID + oo];
    float sg = 1.0f / (1.0f + __expf(-g));
    out[i] = f2bf(attn[i] * sg);
  }
}

// ---------------------------------------------------------------------------
// Workspace layout (bytes)
// ---------------------------------------------------------------------------
static constexpr size_t OFF_HSB  = 0;                           // hs bf16       16 MB
static constexpr size_t OFF_WQB  = OFF_HSB  + (size_t)MTOK*HID*2;
static constexpr size_t OFF_WKB  = OFF_WQB  + (size_t)QGN*HID*2;
static constexpr size_t OFF_WVB  = OFF_WKB  + (size_t)KVN*HID*2;
static constexpr size_t OFF_WOB  = OFF_WVB  + (size_t)KVN*HID*2;
static constexpr size_t OFF_QG   = OFF_WOB  + (size_t)HID*HID*2;   // qg f32   64 MB
static constexpr size_t OFF_KF   = OFF_QG   + (size_t)MTOK*QGN*4;
static constexpr size_t OFF_VF   = OFF_KF   + (size_t)MTOK*KVN*4;
static constexpr size_t OFF_QBF  = OFF_VF   + (size_t)MTOK*KVN*4;  // (b,h,s,d) bf16
static constexpr size_t OFF_KBF  = OFF_QBF  + (size_t)BB*NH*SS*HD*2;
static constexpr size_t OFF_VTB  = OFF_KBF  + (size_t)BB*NKV*SS*HD*2;
static constexpr size_t OFF_ATTF = OFF_VTB  + (size_t)BB*NKV*HD*SS*2;
static constexpr size_t OFF_ATTB = OFF_ATTF + (size_t)MTOK*HID*4;

extern "C" void kernel_launch(void* const* d_in, const int* in_sizes, int n_in,
                              void* d_out, int out_size, void* d_ws, size_t ws_size,
                              hipStream_t stream) {
  const float* hs = (const float*)d_in[0];
  const float* cs = (const float*)d_in[1];
  const float* sn = (const float*)d_in[2];
  const float* Wq = (const float*)d_in[3];
  const float* Wk = (const float*)d_in[4];
  const float* Wv = (const float*)d_in[5];
  const float* Wo = (const float*)d_in[6];
  const float* qw = (const float*)d_in[7];
  const float* kw = (const float*)d_in[8];
  (void)in_sizes; (void)n_in; (void)out_size; (void)ws_size;

  char* ws = (char*)d_ws;
  u16*   hsb  = (u16*)  (ws + OFF_HSB);
  u16*   wqb  = (u16*)  (ws + OFF_WQB);
  u16*   wkb  = (u16*)  (ws + OFF_WKB);
  u16*   wvb  = (u16*)  (ws + OFF_WVB);
  u16*   wob  = (u16*)  (ws + OFF_WOB);
  float* qgf  = (float*)(ws + OFF_QG);
  float* kff  = (float*)(ws + OFF_KF);
  float* vff  = (float*)(ws + OFF_VF);
  u16*   qbf  = (u16*)  (ws + OFF_QBF);
  u16*   kbf  = (u16*)  (ws + OFF_KBF);
  u16*   vtb  = (u16*)  (ws + OFF_VTB);
  float* attf = (float*)(ws + OFF_ATTF);
  u16*   attb = (u16*)  (ws + OFF_ATTB);

  // 1) casts to bf16
  cast_kernel<<<2048, 256, 0, stream>>>(hs, hsb, (size_t)MTOK * HID);
  cast_kernel<<<2048, 256, 0, stream>>>(Wq, wqb, (size_t)QGN  * HID);
  cast_kernel<<<512,  256, 0, stream>>>(Wk, wkb, (size_t)KVN  * HID);
  cast_kernel<<<512,  256, 0, stream>>>(Wv, wvb, (size_t)KVN  * HID);
  cast_kernel<<<1024, 256, 0, stream>>>(Wo, wob, (size_t)HID  * HID);

  // 2) projections (WMMA, async-LDS double buffered)
  gemm_kernel<<<dim3(QGN / 128, MTOK / 128), 256, 0, stream>>>(hsb, wqb, qgf, MTOK, QGN, HID);
  gemm_kernel<<<dim3(KVN / 128, MTOK / 128), 256, 0, stream>>>(hsb, wkb, kff, MTOK, KVN, HID);
  gemm_kernel<<<dim3(KVN / 128, MTOK / 128), 256, 0, stream>>>(hsb, wvb, vff, MTOK, KVN, HID);

  // 3) RMSNorm + RoPE + layout repack (V transposed for PV WMMA)
  normrope_kernel<<<MTOK, 128, 0, stream>>>(qgf, kff, vff, cs, sn, qw, kw, qbf, kbf, vtb);

  // 4) causal flash attention (WMMA)
  attn_kernel<<<dim3(BB * NH, SS / 64), 128, 0, stream>>>(qbf, kbf, vtb, attf);

  // 5) gating
  gate_kernel<<<2048, 256, 0, stream>>>(attf, qgf, attb);

  // 6) output projection (WMMA) straight into d_out (f32)
  gemm_kernel<<<dim3(HID / 128, MTOK / 128), 256, 0, stream>>>(attb, wob, (float*)d_out, MTOK, HID, HID);
}
